// PropertyFocusedAttention_11768210391336
// MI455X (gfx1250) — matrix operand order repaired and checked by reading, not compiled, not run on hardware
//
#include <hip/hip_runtime.h>

// ---------------------------------------------------------------------------
// MHA forward for MI455X (gfx1250, wave32, WMMA bf16 16x16x32, f32 accum)
// B=2, S=2048, D=1024, H=16, Dh=64.
// GEMMs: software-pipelined 32x64 wave tiles.
// Attention: per-block K/V tiles staged in LDS by the Tensor Data Mover
// (tensor_load_to_lds, double buffered, hardware pad for conflict-free banks).
// ---------------------------------------------------------------------------

#define BATCH   2
#define SEQ     2048
#define DMODEL  1024
#define NHEADS  16
#define HDIM    64
#define MTOT    (BATCH * SEQ)          // 4096 rows

typedef __bf16 bf16_t;
typedef __attribute__((ext_vector_type(16))) __bf16 v16bf;
typedef __attribute__((ext_vector_type(8)))  __bf16 v8bf;
typedef __attribute__((ext_vector_type(8)))  float  v8f;
typedef __attribute__((ext_vector_type(4))) unsigned u32x4;
typedef __attribute__((ext_vector_type(8))) int      i32x8;
typedef __attribute__((ext_vector_type(4))) int      i32x4;

union AB16 { v16bf v; v8bf h[2]; };

// A fragment, 16x32 bf16 (MxK), row-major source, leading dim ld (elements).
// lane = 16*half + r holds row r; v0-3 = K[8*half..+7], v4-7 = K[16+8*half..+7].
__device__ __forceinline__ v16bf load_a_frag(const bf16_t* __restrict__ A,
                                             int ld, int m0, int k0, int lane) {
  int hf = lane >> 4, r = lane & 15;
  const bf16_t* p = A + (size_t)(m0 + r) * ld;
  AB16 u;
  u.h[0] = *(const v8bf*)(p + k0 + 8 * hf);
  u.h[1] = *(const v8bf*)(p + k0 + 16 + 8 * hf);
  return u.v;
}

// B fragment, 32x16 bf16 (KxN), memory laid out Bt[n][k] (B column contiguous
// in k), leading dim ld.  Lane = 16*half + r -> column n0+r, K[k0+8*half..+7], K[k0+16+8*half..+7].
__device__ __forceinline__ v16bf load_b_frag(const bf16_t* __restrict__ Bt,
                                             int ld, int n0, int k0, int lane) {
  int hf = lane >> 4, r = lane & 15;
  const bf16_t* p = Bt + (size_t)(n0 + r) * ld + k0 + 8 * hf;
  AB16 u;
  u.h[0] = *(const v8bf*)(p);
  u.h[1] = *(const v8bf*)(p + 16);
  return u.v;
}

// Same B fragment but from an LDS tile (emits ds_load_b128 pairs).
__device__ __forceinline__ v16bf lds_b_frag(const bf16_t* base, int ld,
                                            int n0, int k0, int lane) {
  int hf = lane >> 4, r = lane & 15;
  const bf16_t* p = base + (n0 + r) * ld + k0 + 8 * hf;
  AB16 u;
  u.h[0] = *(const v8bf*)(p);
  u.h[1] = *(const v8bf*)(p + 16);
  return u.v;
}

__device__ __forceinline__ v8f wmma_bf16(v16bf a, v16bf b, v8f c) {
  return __builtin_amdgcn_wmma_f32_16x16x32_bf16(false, a, false, b,
                                                 (short)0, c, false, false);
}

// ---------------------------------------------------------------------------
// Tensor Data Mover: 2D tile (data_size = 8B units) global -> LDS with
// hardware row padding.  D# built per ISA cdna5 8.3/8.4.
// ---------------------------------------------------------------------------
__device__ __forceinline__ void tdm_load_2d(unsigned lds_off, const void* gptr,
                                            unsigned tensor_d0, unsigned tensor_d1,
                                            unsigned tile_d0, unsigned tile_d1,
                                            unsigned stride0,
                                            unsigned pad_int_code,
                                            unsigned pad_amt_code) {
  unsigned long long ga = (unsigned long long)(size_t)gptr;
  u32x4 g0;
  g0[0] = 1u;                                    // count=1, user mode
  g0[1] = lds_off;                               // LDS byte address
  g0[2] = (unsigned)ga;                          // global_addr[31:0]
  g0[3] = (unsigned)((ga >> 32) & 0x1FFFFFFu) | (2u << 30);  // [56:32] | type=2
  i32x8 g1;
  g1[0] = (int)((3u << 16) |                     // data_size = 8B
                (1u << 20) |                     // pad_enable
                (pad_int_code << 22) | (pad_amt_code << 25));
  g1[1] = (int)((tensor_d0 & 0xFFFFu) << 16);    // tensor_dim0[15:0]
  g1[2] = (int)((tensor_d0 >> 16) | ((tensor_d1 & 0xFFFFu) << 16));
  g1[3] = (int)((tensor_d1 >> 16) | (tile_d0 << 16));
  g1[4] = (int)tile_d1;                          // tile_dim1 | tile_dim2=0
  g1[5] = (int)stride0;                          // tensor_dim0_stride[31:0]
  g1[6] = 0;
  g1[7] = 0;
  i32x4 z4 = {0, 0, 0, 0};
#if __clang_major__ >= 23
  i32x8 z8 = {0, 0, 0, 0, 0, 0, 0, 0};
  __builtin_amdgcn_tensor_load_to_lds(g0, g1, z4, z4, z8, 0);
#else
  __builtin_amdgcn_tensor_load_to_lds(g0, g1, z4, z4, 0);
#endif
}

// ---------------------------------------------------------------------------
// fp32 -> bf16 conversion (4 elements / thread)
// ---------------------------------------------------------------------------
__global__ __launch_bounds__(256) void cvt_kernel(const float* __restrict__ src,
                                                  bf16_t* __restrict__ dst, int n) {
  int i = (blockIdx.x * blockDim.x + threadIdx.x) * 4;
  if (i + 3 < n) {
    float4 v = *(const float4*)(src + i);
    dst[i + 0] = (bf16_t)v.x;
    dst[i + 1] = (bf16_t)v.y;
    dst[i + 2] = (bf16_t)v.z;
    dst[i + 3] = (bf16_t)v.w;
  }
}

// ---------------------------------------------------------------------------
// Software-pipelined 32x64 GEMM tile core: acc += X[m0:m0+32, :] * W^T[:, n0:n0+64]
// ---------------------------------------------------------------------------
__device__ __forceinline__ void gemm_core_32x64(const bf16_t* __restrict__ X,
                                                const bf16_t* __restrict__ W,
                                                int m0, int n0, int lane,
                                                v8f acc[2][4]) {
  v16bf a0 = load_a_frag(X, DMODEL, m0, 0, lane);
  v16bf a1 = load_a_frag(X, DMODEL, m0 + 16, 0, lane);
  v16bf b0 = load_b_frag(W, DMODEL, n0, 0, lane);
  v16bf b1 = load_b_frag(W, DMODEL, n0 + 16, 0, lane);
  v16bf b2 = load_b_frag(W, DMODEL, n0 + 32, 0, lane);
  v16bf b3 = load_b_frag(W, DMODEL, n0 + 48, 0, lane);
  for (int k = 32; k < DMODEL; k += 32) {
    // prefetch next k-step while current WMMAs execute
    v16bf na0 = load_a_frag(X, DMODEL, m0, k, lane);
    v16bf na1 = load_a_frag(X, DMODEL, m0 + 16, k, lane);
    v16bf nb0 = load_b_frag(W, DMODEL, n0, k, lane);
    v16bf nb1 = load_b_frag(W, DMODEL, n0 + 16, k, lane);
    v16bf nb2 = load_b_frag(W, DMODEL, n0 + 32, k, lane);
    v16bf nb3 = load_b_frag(W, DMODEL, n0 + 48, k, lane);
    acc[0][0] = wmma_bf16(a0, b0, acc[0][0]);
    acc[0][1] = wmma_bf16(a0, b1, acc[0][1]);
    acc[0][2] = wmma_bf16(a0, b2, acc[0][2]);
    acc[0][3] = wmma_bf16(a0, b3, acc[0][3]);
    acc[1][0] = wmma_bf16(a1, b0, acc[1][0]);
    acc[1][1] = wmma_bf16(a1, b1, acc[1][1]);
    acc[1][2] = wmma_bf16(a1, b2, acc[1][2]);
    acc[1][3] = wmma_bf16(a1, b3, acc[1][3]);
    a0 = na0; a1 = na1; b0 = nb0; b1 = nb1; b2 = nb2; b3 = nb3;
  }
  acc[0][0] = wmma_bf16(a0, b0, acc[0][0]);
  acc[0][1] = wmma_bf16(a0, b1, acc[0][1]);
  acc[0][2] = wmma_bf16(a0, b2, acc[0][2]);
  acc[0][3] = wmma_bf16(a0, b3, acc[0][3]);
  acc[1][0] = wmma_bf16(a1, b0, acc[1][0]);
  acc[1][1] = wmma_bf16(a1, b1, acc[1][1]);
  acc[1][2] = wmma_bf16(a1, b2, acc[1][2]);
  acc[1][3] = wmma_bf16(a1, b3, acc[1][3]);
}

// ---------------------------------------------------------------------------
// QKV projection.  mode 0: bf16 [(b*H+h)*S+s]*64+d (Q,K);  mode 1: V transposed.
// Wave tile 32x64; 2048 waves.
// ---------------------------------------------------------------------------
__global__ __launch_bounds__(256) void gemm_qkv_kernel(
    const bf16_t* __restrict__ X, const bf16_t* __restrict__ W,
    const float* __restrict__ bias, bf16_t* __restrict__ dst, int mode) {
  int lane = threadIdx.x & 31;
  int w = blockIdx.x * 8 + (threadIdx.x >> 5);
  int m0 = (w >> 4) << 5;                    // 128 strips of 32 rows
  int n0 = (w & 15) << 6;                    // 16 strips of 64 cols
  v8f acc[2][4] = {};
  gemm_core_32x64(X, W, m0, n0, lane, acc);
  int hf = lane >> 4, c = lane & 15;
#pragma unroll
  for (int t = 0; t < 4; ++t) {
    int n = n0 + 16 * t + c;
    float bv = bias[n];
    int h = n >> 6, d = n & 63;
#pragma unroll
    for (int mm = 0; mm < 2; ++mm)
#pragma unroll
      for (int i = 0; i < 8; ++i) {
        int m = m0 + 16 * mm + i + 8 * hf;   // C layout: M = i + 8*(lane/16)
        int bb = m >> 11, s = m & (SEQ - 1);
        float val = acc[mm][t][i] + bv;
        size_t idx = (mode == 0)
            ? ((size_t)((bb * NHEADS + h) * SEQ + s)) * HDIM + d
            : ((size_t)((bb * NHEADS + h) * HDIM + d)) * SEQ + s;
        dst[idx] = (bf16_t)val;
      }
  }
}

// ---------------------------------------------------------------------------
// Flash attention.  Block = 8 waves, one (b,h), 128 query rows (16/wave).
// K tile (32x64) and V^T tile (64x32) staged to LDS by TDM, double buffered,
// hardware-padded rows: K stride 72 elems (144B), V stride 40 elems (80B)
// -> conflict-free 16B ds_load fragments.
// ---------------------------------------------------------------------------
__global__ __launch_bounds__(256) void attn_kernel(
    const bf16_t* __restrict__ Q, const bf16_t* __restrict__ K,
    const bf16_t* __restrict__ VT, bf16_t* __restrict__ AO) {
  __shared__ __attribute__((aligned(128))) bf16_t Kt[2][32 * 72];
  __shared__ __attribute__((aligned(128))) bf16_t Vt[2][64 * 40];
  __shared__ __attribute__((aligned(128))) bf16_t Pl[8][16 * 40];

  int lane = threadIdx.x & 31;
  int wv = threadIdx.x >> 5;
  int bh = blockIdx.x >> 4;                       // 0..31 (b*16+h)
  int q0 = ((blockIdx.x & 15) << 7) + (wv << 4);  // this wave's query tile
  const bf16_t* Qh = Q  + (size_t)bh * SEQ * HDIM;
  const bf16_t* Kh = K  + (size_t)bh * SEQ * HDIM;
  const bf16_t* Vh = VT + (size_t)bh * HDIM * SEQ;

  v16bf qa0 = load_a_frag(Qh, HDIM, q0, 0, lane);
  v16bf qa1 = load_a_frag(Qh, HDIM, q0, 32, lane);

  v8f acc[4] = {};
  float mrow[8], lrow[8];
#pragma unroll
  for (int i = 0; i < 8; ++i) { mrow[i] = -3.0e38f; lrow[i] = 0.f; }
  const float sc = 0.125f * 1.44269504089f;       // (1/sqrt(64)) * log2(e)
  int hf = lane >> 4, c = lane & 15;
  bf16_t* pl = &Pl[wv][0];

  // Prologue: wave 0 DMAs the first K/V tiles into buffer 0.
  if (threadIdx.x < 32) {
    // K tile: rows 32, row = 64 elems = 16x8B; pad 16B per 128B (codes 4,3)
    tdm_load_2d((unsigned)(size_t)&Kt[0][0], Kh, 16u, SEQ, 16u, 32u, 16u, 4u, 3u);
    // V tile: rows 64 (d), row = 32 elems = 8x8B; pad 16B per 64B (codes 3,3)
    tdm_load_2d((unsigned)(size_t)&Vt[0][0], Vh, 512u, HDIM, 8u, 64u, 512u, 3u, 3u);
  }

  for (int it = 0; it < SEQ / 32; ++it) {
    int cur = it & 1;
    if (threadIdx.x < 32) __builtin_amdgcn_s_wait_tensorcnt(0);
    __syncthreads();                               // tile `cur` visible to all
    if (it + 1 < SEQ / 32 && threadIdx.x < 32) {   // DMA next tile, overlapped
      int j1 = (it + 1) * 32;
      tdm_load_2d((unsigned)(size_t)&Kt[cur ^ 1][0], Kh + (size_t)j1 * HDIM,
                  16u, SEQ, 16u, 32u, 16u, 4u, 3u);
      tdm_load_2d((unsigned)(size_t)&Vt[cur ^ 1][0], Vh + j1,
                  512u, HDIM, 8u, 64u, 512u, 3u, 3u);
    }
    const bf16_t* kt = &Kt[cur][0];
    const bf16_t* vt = &Vt[cur][0];

    // ---- scores: 16x32 tile as two 16x16 C tiles (from LDS K tile) --------
    v8f s0 = {}, s1 = {};
    {
      v16bf b;
      b = lds_b_frag(kt, 72, 0, 0, lane);   s0 = wmma_bf16(qa0, b, s0);
      b = lds_b_frag(kt, 72, 0, 32, lane);  s0 = wmma_bf16(qa1, b, s0);
      b = lds_b_frag(kt, 72, 16, 0, lane);  s1 = wmma_bf16(qa0, b, s1);
      b = lds_b_frag(kt, 72, 16, 32, lane); s1 = wmma_bf16(qa1, b, s1);
    }
    // ---- online softmax in exp2 domain ------------------------------------
    float f[8];
#pragma unroll
    for (int i = 0; i < 8; ++i) {
      float a0 = s0[i] * sc, a1 = s1[i] * sc;
      s0[i] = a0; s1[i] = a1;
      float mx = fmaxf(a0, a1);
#pragma unroll
      for (int off = 8; off >= 1; off >>= 1)
        mx = fmaxf(mx, __shfl_xor(mx, off, 32));   // 16-lane row reduction
      float nm = fmaxf(mrow[i], mx);
      f[i] = exp2f(mrow[i] - nm);
      mrow[i] = nm;
    }
#pragma unroll
    for (int i = 0; i < 8; ++i) {
      float p0 = exp2f(s0[i] - mrow[i]);
      float p1 = exp2f(s1[i] - mrow[i]);
      s0[i] = p0; s1[i] = p1;
      float rs = p0 + p1;
#pragma unroll
      for (int off = 8; off >= 1; off >>= 1)
        rs += __shfl_xor(rs, off, 32);
      lrow[i] = lrow[i] * f[i] + rs;
    }
#pragma unroll
    for (int t = 0; t < 4; ++t)
#pragma unroll
      for (int i = 0; i < 8; ++i) acc[t][i] *= f[i];

    // ---- P: C-layout -> LDS (row-major 16x32, stride 40) -> A-layout ------
#pragma unroll
    for (int i = 0; i < 8; ++i) {
      int row = i + 8 * hf;
      pl[row * 40 + c]      = (bf16_t)s0[i];
      pl[row * 40 + 16 + c] = (bf16_t)s1[i];
    }
    asm volatile("s_wait_dscnt 0" ::: "memory");   // wave-private RAW guard
    AB16 u;
    const bf16_t* pr = pl + (lane & 15) * 40;
    u.h[0] = *(const v8bf*)(pr + 8 * hf);
    u.h[1] = *(const v8bf*)(pr + 16 + 8 * hf);
    v16bf pa = u.v;

    // ---- P x V from LDS V tile --------------------------------------------
#pragma unroll
    for (int t = 0; t < 4; ++t) {
      v16bf b = lds_b_frag(vt, 40, 16 * t, 0, lane);
      acc[t] = wmma_bf16(pa, b, acc[t]);
    }
  }

  // ---- finalize: divide by row sum, store bf16 [m][DMODEL] ----------------
  int bb = bh >> 4, h = bh & 15;
#pragma unroll
  for (int t = 0; t < 4; ++t)
#pragma unroll
    for (int i = 0; i < 8; ++i) {
      int m = (bb << 11) + q0 + i + 8 * hf;
      int n = h * HDIM + t * 16 + c;
      float val = acc[t][i] / lrow[i];
      AO[(size_t)m * DMODEL + n] = (bf16_t)val;
    }
}

// ---------------------------------------------------------------------------
// Output projection: out[m][n] = sum_k AO[m][k] * Wo[n][k] + bo[n]  (fp32 out)
// ---------------------------------------------------------------------------
__global__ __launch_bounds__(256) void out_proj_kernel(
    const bf16_t* __restrict__ A, const bf16_t* __restrict__ W,
    const float* __restrict__ bias, float* __restrict__ out) {
  int lane = threadIdx.x & 31;
  int w = blockIdx.x * 8 + (threadIdx.x >> 5);
  int m0 = (w >> 4) << 5;
  int n0 = (w & 15) << 6;
  v8f acc[2][4] = {};
  gemm_core_32x64(A, W, m0, n0, lane, acc);
  int hf = lane >> 4, c = lane & 15;
#pragma unroll
  for (int t = 0; t < 4; ++t) {
    int n = n0 + 16 * t + c;
    float bv = bias[n];
#pragma unroll
    for (int mm = 0; mm < 2; ++mm)
#pragma unroll
      for (int i = 0; i < 8; ++i) {
        int m = m0 + 16 * mm + i + 8 * hf;
        out[(size_t)m * DMODEL + n] = acc[mm][t][i] + bv;
      }
  }
}

// ---------------------------------------------------------------------------
// Workspace layout (bytes):
//   0   xb   bf16 [4096,1024]  8MB | 8M Wqb 2MB | 10M Wkb | 12M Wvb | 14M Wob
//  16M  qb [B,H,S,64] 8MB | 24M kb 8MB | 32M vTb [B,H,64,S] 8MB | 40M aob 8MB
//  Total 48 MB (entirely L2-resident on the 192MB L2).
// ---------------------------------------------------------------------------
extern "C" void kernel_launch(void* const* d_in, const int* in_sizes, int n_in,
                              void* d_out, int out_size, void* d_ws, size_t ws_size,
                              hipStream_t stream) {
  const float* x  = (const float*)d_in[0];
  const float* Wq = (const float*)d_in[1];
  const float* bq = (const float*)d_in[2];
  const float* Wk = (const float*)d_in[3];
  const float* bk = (const float*)d_in[4];
  const float* Wv = (const float*)d_in[5];
  const float* bv = (const float*)d_in[6];
  const float* Wo = (const float*)d_in[7];
  const float* bo = (const float*)d_in[8];
  float* out = (float*)d_out;

  char* ws = (char*)d_ws;
  bf16_t* xb  = (bf16_t*)(ws + 0);
  bf16_t* Wqb = (bf16_t*)(ws + ((size_t)8  << 20));
  bf16_t* Wkb = (bf16_t*)(ws + ((size_t)10 << 20));
  bf16_t* Wvb = (bf16_t*)(ws + ((size_t)12 << 20));
  bf16_t* Wob = (bf16_t*)(ws + ((size_t)14 << 20));
  bf16_t* qb  = (bf16_t*)(ws + ((size_t)16 << 20));
  bf16_t* kb  = (bf16_t*)(ws + ((size_t)24 << 20));
  bf16_t* vTb = (bf16_t*)(ws + ((size_t)32 << 20));
  bf16_t* aob = (bf16_t*)(ws + ((size_t)40 << 20));

  cvt_kernel<<<4096, 256, 0, stream>>>(x,  xb,  MTOT * DMODEL);
  cvt_kernel<<<1024, 256, 0, stream>>>(Wq, Wqb, DMODEL * DMODEL);
  cvt_kernel<<<1024, 256, 0, stream>>>(Wk, Wkb, DMODEL * DMODEL);
  cvt_kernel<<<1024, 256, 0, stream>>>(Wv, Wvb, DMODEL * DMODEL);
  cvt_kernel<<<1024, 256, 0, stream>>>(Wo, Wob, DMODEL * DMODEL);

  gemm_qkv_kernel<<<256, 256, 0, stream>>>(xb, Wqb, bq, qb,  0);
  gemm_qkv_kernel<<<256, 256, 0, stream>>>(xb, Wkb, bk, kb,  0);
  gemm_qkv_kernel<<<256, 256, 0, stream>>>(xb, Wvb, bv, vTb, 1);

  attn_kernel<<<512, 256, 0, stream>>>(qb, kb, vTb, aob);

  out_proj_kernel<<<256, 256, 0, stream>>>(aob, Wob, bo, out);
}
